// FastMVSNet_pipe_46858093199612
// MI455X (gfx1250) — matrix operand order repaired and checked by reading, not compiled, not run on hardware
//
#include <hip/hip_runtime.h>
#include <hip/hip_bf16.h>

// ---------------------------------------------------------------------------
// FastMVSNet forward for MI455X (gfx1250).
// Heavy convs = implicit GEMM on v_wmma_f32_16x16x32_f16 (wave32). Each wave
// owns a 16(Cout) x 64(pixel) tile: one A (weight) panel feeds 4 WMMAs per
// K-chunk, amortizing weight staging and A-fragment loads 4x.
// Cost volume is computed fused (grid-sample + variance) and stored f16 to
// halve HBM traffic into the 3D UNet.
// ---------------------------------------------------------------------------

typedef __attribute__((ext_vector_type(16))) _Float16 v16h;
typedef __attribute__((ext_vector_type(8)))  float    v8f;

#define B_    1
#define V_    3
#define HIMG  384
#define WIMG  512
#define DDEP  32
#define CF    32
#define HF    96
#define WF    128
#define RS    4
#define EPSF  1e-8f
#define NT    4              // pixel groups of 16 per wave (N-tile = 64)

static __device__ __forceinline__ int iclamp(int v, int lo, int hi) {
  return v < lo ? lo : (v > hi ? hi : v);
}

// ---------------------------------------------------------------------------
// Generic implicit-GEMM conv (2D: ID=KD=1,padD=0; 3D; transposed 3D with
// lhs_dilation=2, pad lo=1). One wave per 16(Cout) x 64(pixel) output tile.
// D = A(16x32 weights) x B(32x16 patches) + C, accumulated f32, 4 N-groups.
// Epilogue: optional BN scale/bias + ReLU.
// ---------------------------------------------------------------------------
__global__ __launch_bounds__(32) void conv_wmma_kernel(
    const void* __restrict__ inp, int inHalf,
    const float* __restrict__ wgt,
    const float* __restrict__ gamma, const float* __restrict__ beta,
    float* __restrict__ out,
    int Cin, int Cout,
    int ID, int IH, int IW,
    int OD, int OH, int OW,
    int KD, int KH, int KW,
    int stride, int padD, int padHW,
    int transposed, int relu)
{
  __shared__ _Float16 sA[16 * 32];        // [m][k] weight panel
  __shared__ _Float16 sB[32 * (16 * NT)]; // [k][n] patch panel (64 pixels)

  const int lane    = threadIdx.x;
  const int NPIX    = OD * OH * OW;
  const int pixBase = blockIdx.x * (16 * NT);
  const int coBase  = blockIdx.y * 16;
  const int z       = blockIdx.z;
  const int KT      = KD * KH * KW;
  const int Ktot    = Cin * KT;

  const float*    inF = (const float*)inp;
  const _Float16* inH = (const _Float16*)inp;
  const size_t inBase = (size_t)z * Cin * ID * IH * IW;

  // Packed output coords of this tile's 64 pixels:
  // bit31 = invalid, d in [25:20], h in [19:10], w in [9:0].
  int pc[16 * NT];
#pragma unroll 8
  for (int n = 0; n < 16 * NT; n++) {
    int p = pixBase + n;
    if (p < NPIX) {
      int d = p / (OH * OW);
      int r = p - d * OH * OW;
      int h = r / OW;
      int w = r - h * OW;
      pc[n] = w | (h << 10) | (d << 20);
    } else {
      pc[n] = (int)0x80000000;
    }
  }

  const int mRow = lane >> 1;   // A staging: 2 lanes per weight row
  const int kOff = (lane & 1) * 16;

  v8f acc[NT];
#pragma unroll
  for (int g = 0; g < NT; g++) acc[g] = (v8f){};

  for (int k0 = 0; k0 < Ktot; k0 += 32) {
    // ---- stage A: 16 output channels x 32 reduction slots of weights ----
    {
      const bool mok = (coBase + mRow) < Cout;
      const float* wrow = wgt + (size_t)(coBase + mRow) * Ktot + k0 + kOff;
      __builtin_prefetch(wrow + 32, 0, 0);   // global_prefetch_b8 (next panel)
#pragma unroll
      for (int i = 0; i < 16; i++) {
        int kk = k0 + kOff + i;
        sA[mRow * 32 + kOff + i] =
            (mok && kk < Ktot) ? (_Float16)wrow[i] : (_Float16)0.0f;
      }
    }

    // ---- stage B: one reduction row per lane, 64 pixels wide ----
    {
      int kr = k0 + lane;
      bool krok = kr < Ktot;
      int ci = 0, kd = 0, kh = 0, kw = 0;
      if (krok) {
        ci = kr / KT;
        int tap = kr - ci * KT;
        kd = tap / (KH * KW);
        int r2 = tap - kd * KH * KW;
        kh = r2 / KW;
        kw = r2 - kh * KW;
      }
#pragma unroll 8
      for (int n = 0; n < 16 * NT; n++) {
        float v = 0.0f;
        int pcv = pc[n];
        if (krok && pcv >= 0) {
          int pwv = pcv & 1023;
          int phv = (pcv >> 10) & 1023;
          int pdv = (pcv >> 20) & 63;
          int id, ih, iw;
          bool ok;
          if (!transposed) {
            id = pdv * stride + kd - padD;
            ih = phv * stride + kh - padHW;
            iw = pwv * stride + kw - padHW;
            ok = (id >= 0) & (id < ID) & (ih >= 0) & (ih < IH) &
                 (iw >= 0) & (iw < IW);
          } else {
            // lhs_dilation=2, pad lo=1 (deconv3d): src = (o + k - 1)/2 if even
            int jd = pdv + kd - 1, jh = phv + kh - 1, jw = pwv + kw - 1;
            ok = (jd >= 0) & (jh >= 0) & (jw >= 0) &
                 ((jd & 1) == 0) & ((jh & 1) == 0) & ((jw & 1) == 0);
            id = jd >> 1; ih = jh >> 1; iw = jw >> 1;
            ok = ok & (id < ID) & (ih < IH) & (iw < IW);
          }
          if (ok) {
            size_t idx = inBase + (((size_t)ci * ID + id) * IH + ih) * IW + iw;
            if (inHalf) v = (float)inH[idx];
            else        v = inF[idx];
          }
        }
        sB[lane * (16 * NT) + n] = (_Float16)v;
      }
    }
    __syncthreads();

    // ---- load fragments per CDNA5 wave32 WMMA layouts, 4 WMMAs / chunk ----
    const int mA   = lane & 15;
    const int half = lane >> 4;
    v16h a;
#pragma unroll
    for (int r = 0; r < 8; r++) {
      int kb = (r < 4) ? (2 * r + 8 * half) : (16 + 2 * (r - 4) + 8 * half);
      a[2 * r]     = sA[mA * 32 + kb];
      a[2 * r + 1] = sA[mA * 32 + kb + 1];
    }
#pragma unroll
    for (int g = 0; g < NT; g++) {
      v16h b;
#pragma unroll
      for (int r = 0; r < 8; r++) {
        int k0b = (r < 4) ? (2 * r + 8 * half) : (16 + 2 * (r - 4) + 8 * half);
        b[2 * r]     = sB[k0b * (16 * NT) + g * 16 + mA];
        b[2 * r + 1] = sB[(k0b + 1) * (16 * NT) + g * 16 + mA];
      }
      acc[g] = __builtin_amdgcn_wmma_f32_16x16x32_f16(
          /*neg_a=*/false, a, /*neg_b=*/false, b,
          /*c_mod=*/(short)0, acc[g], /*reuse_a=*/false, /*reuse_b=*/false);
    }
    __syncthreads();
  }

  // ---- epilogue: BN + ReLU + store (C layout: m = r + 8*half, n = lane&15)
  const int nSt  = lane & 15;
  const int hfSt = lane >> 4;
#pragma unroll
  for (int g = 0; g < NT; g++) {
#pragma unroll
    for (int r = 0; r < 8; r++) {
      int m  = r + 8 * hfSt;
      int co = coBase + m;
      int p  = pixBase + g * 16 + nSt;
      if (co < Cout && p < NPIX) {
        float v = acc[g][r];
        if (gamma) v = v * gamma[co] + beta[co];
        if (relu)  v = v > 0.0f ? v : 0.0f;
        out[(size_t)z * Cout * NPIX + (size_t)co * NPIX + p] = v;
      }
    }
  }
}

// ---------------------------------------------------------------------------
// Strided spatial downsample: out[nc,ho,wo] = in[nc, ho*s, wo*s]
// ---------------------------------------------------------------------------
__global__ __launch_bounds__(256) void downsample_kernel(
    const float* __restrict__ in, float* __restrict__ out,
    int NC, int H, int W, int s, int Ho, int Wo)
{
  int t = blockIdx.x * blockDim.x + threadIdx.x;
  int total = NC * Ho * Wo;
  if (t >= total) return;
  int wo = t % Wo;
  int ho = (t / Wo) % Ho;
  int nc = t / (Wo * Ho);
  out[t] = in[((size_t)nc * H + (size_t)ho * s) * W + (size_t)wo * s];
}

// ---------------------------------------------------------------------------
// Fused homography grid-sample + variance cost. Avoids materializing the
// 453 MB `pf` tensor; feature maps (4.7 MB) stay L2-resident.
// cost[(i*CF+c), d, h, w] in f16, layout (B*V, CF, D, HF, WF).
// ---------------------------------------------------------------------------
__global__ __launch_bounds__(256) void cost_fuse_kernel(
    const float* __restrict__ feat,   // (V, CF, HF, WF)
    const float* __restrict__ bp,     // (V, V, 3, HF, WF) downsampled
    const float* __restrict__ dr,     // (V, V, 3, HF, WF)
    const float* __restrict__ dd,     // (V, D, HF, WF)
    _Float16* __restrict__ cost)
{
  int t = blockIdx.x * blockDim.x + threadIdx.x;
  const int total = V_ * DDEP * HF * WF;
  if (t >= total) return;
  int w = t % WF;
  int h = (t / WF) % HF;
  int d = (t / (WF * HF)) % DDEP;
  int i = t / (WF * HF * DDEP);

  float s1[CF], s2[CF];
#pragma unroll
  for (int c = 0; c < CF; c++) { s1[c] = 0.0f; s2[c] = 0.0f; }

  const float ddv = dd[(((size_t)i * DDEP + d) * HF + h) * WF + w];
  const size_t chs = (size_t)HF * WF;

  for (int j = 0; j < V_; j++) {
    size_t gb = ((((size_t)i * V_ + j) * 3) * HF + h) * WF + w;
    float wx = bp[gb] + dr[gb] * ddv;
    float wy = bp[gb + chs] + dr[gb + chs] * ddv;
    float wz = bp[gb + 2 * chs] + dr[gb + 2 * chs] * ddv + EPSF;
    // grid = wg.xy / wg.z - 1 ;  x = (g+1)*W/2 - 0.5
    float x = (wx / wz) * (WF * 0.5f) - 0.5f;
    float y = (wy / wz) * (HF * 0.5f) - 0.5f;
    float x0f = floorf(x), y0f = floorf(y);
    float fx = x - x0f, fy = y - y0f;
    int x0 = (int)x0f, y0 = (int)y0f;
    int x1 = x0 + 1,   y1 = y0 + 1;
    bool vx0 = (x0 >= 0) & (x0 < WF), vx1 = (x1 >= 0) & (x1 < WF);
    bool vy0 = (y0 >= 0) & (y0 < HF), vy1 = (y1 >= 0) & (y1 < HF);
    int cx0 = iclamp(x0, 0, WF - 1), cx1 = iclamp(x1, 0, WF - 1);
    int cy0 = iclamp(y0, 0, HF - 1), cy1 = iclamp(y1, 0, HF - 1);
    float w00 = (1.0f - fx) * (1.0f - fy) * ((vx0 & vy0) ? 1.0f : 0.0f);
    float w10 = fx * (1.0f - fy)          * ((vx1 & vy0) ? 1.0f : 0.0f);
    float w01 = (1.0f - fx) * fy          * ((vx0 & vy1) ? 1.0f : 0.0f);
    float w11 = fx * fy                   * ((vx1 & vy1) ? 1.0f : 0.0f);
    int o00 = cy0 * WF + cx0, o10 = cy0 * WF + cx1;
    int o01 = cy1 * WF + cx0, o11 = cy1 * WF + cx1;
    const float* fj = feat + (size_t)j * CF * chs;
#pragma unroll
    for (int c = 0; c < CF; c++) {
      const float* fc = fj + (size_t)c * chs;
      float v = w00 * fc[o00] + w10 * fc[o10] + w01 * fc[o01] + w11 * fc[o11];
      s1[c] += v;
      s2[c] += v * v;
    }
  }

  const float invV = 1.0f / (float)V_;
  size_t ob = (size_t)i * CF * DDEP * chs + (size_t)d * chs +
              (size_t)h * WF + w;
#pragma unroll
  for (int c = 0; c < CF; c++) {
    float mu = s1[c] * invV;
    cost[ob + (size_t)c * DDEP * chs] = (_Float16)(s2[c] * invV - mu * mu);
  }
}

// ---------------------------------------------------------------------------
// pred = sum_d dd * softmax_d(-fcv)
// ---------------------------------------------------------------------------
__global__ __launch_bounds__(256) void depth_regress_kernel(
    const float* __restrict__ fcv,   // (V, D, HF*WF)
    const float* __restrict__ dd,    // (V, D, HF*WF)
    float* __restrict__ pred)        // (V, HF*WF)
{
  int t = blockIdx.x * blockDim.x + threadIdx.x;
  const int HW = HF * WF;
  if (t >= V_ * HW) return;
  int p = t % HW;
  int i = t / HW;
  float f[DDEP];
  float mx = -3.4e38f;
#pragma unroll
  for (int d = 0; d < DDEP; d++) {
    f[d] = -fcv[((size_t)i * DDEP + d) * HW + p];
    mx = f[d] > mx ? f[d] : mx;
  }
  float s = 0.0f;
#pragma unroll
  for (int d = 0; d < DDEP; d++) { f[d] = __expf(f[d] - mx); s += f[d]; }
  float acc = 0.0f;
#pragma unroll
  for (int d = 0; d < DDEP; d++)
    acc += dd[((size_t)i * DDEP + d) * HW + p] * f[d];
  pred[t] = acc / s;
}

// ---------------------------------------------------------------------------
// Propagation: softmax over 9 logits, blend 3x3 edge-padded depth neighbors.
// ---------------------------------------------------------------------------
__global__ __launch_bounds__(256) void prop_combine_kernel(
    const float* __restrict__ logits,  // (V, 9, HF*WF)
    const float* __restrict__ depth,   // (V, HF*WF)
    float* __restrict__ out)           // (V, HF*WF)
{
  int t = blockIdx.x * blockDim.x + threadIdx.x;
  const int HW = HF * WF;
  if (t >= V_ * HW) return;
  int p = t % HW;
  int i = t / HW;
  int h = p / WF, w = p % WF;
  float l[9];
  float mx = -3.4e38f;
#pragma unroll
  for (int k = 0; k < 9; k++) {
    l[k] = logits[((size_t)i * 9 + k) * HW + p];
    mx = l[k] > mx ? l[k] : mx;
  }
  float s = 0.0f, acc = 0.0f;
#pragma unroll
  for (int k = 0; k < 9; k++) {
    float e = __expf(l[k] - mx);
    s += e;
    int hh = iclamp(h + k / 3 - 1, 0, HF - 1);
    int ww = iclamp(w + k % 3 - 1, 0, WF - 1);
    acc += depth[(size_t)i * HW + hh * WF + ww] * e;
  }
  out[t] = acc / s;
}

__global__ __launch_bounds__(256) void add_inplace_kernel(
    float* __restrict__ dst, const float* __restrict__ src, int n)
{
  int t = blockIdx.x * blockDim.x + threadIdx.x;
  if (t < n) dst[t] += src[t];
}

// ---------------------------------------------------------------------------
// Host orchestration
// ---------------------------------------------------------------------------
extern "C" void kernel_launch(void* const* d_in, const int* in_sizes, int n_in,
                              void* d_out, int out_size, void* d_ws, size_t ws_size,
                              hipStream_t stream) {
  (void)in_sizes; (void)n_in; (void)out_size;

  // ---- param leaf indices (insertion-order dict, tree-flattened params) ----
  const int LI_IMG = 0;                                     // 8 x (w,g,b)
  const int LI_PROP_IMG = 24, LI_W1 = 48, LI_W2 = 51;
  const int LI_C0_1 = 52, LI_C1_0 = 55, LI_C1_1 = 58, LI_C2_0 = 61;
  const int LI_C2_1 = 64, LI_C3_0 = 67, LI_C3_1 = 70, LI_C6_2 = 73;
  const int LI_D4 = 74, LI_D5 = 77, LI_D6 = 80;
  const int IN_BP = 83, IN_DR = 84, IN_IMG = 85, IN_DEP = 86;

  auto F = [&](int i) -> const float* { return (const float*)d_in[i]; };

  // ---- workspace bump allocator ----
  char* wsp = (char*)d_ws;
  size_t off = 0;
  auto alloc = [&](size_t bytes) -> void* {
    void* p = wsp + off;
    off = (off + bytes + 255) & ~(size_t)255;
    return p;
  };

  float* bufA = (float*)alloc((size_t)3 * 8 * HIMG * WIMG * 4);
  float* bufB = (float*)alloc((size_t)3 * 8 * HIMG * WIMG * 4);
  float* feat = (float*)alloc((size_t)3 * CF * HF * WF * 4);
  float* bpds = (float*)alloc((size_t)V_ * V_ * 3 * HF * WF * 4);
  float* drds = (float*)alloc((size_t)V_ * V_ * 3 * HF * WF * 4);
  float* ddds = (float*)alloc((size_t)V_ * DDEP * HF * WF * 4);
  _Float16* cost = (_Float16*)alloc((size_t)3 * CF * DDEP * HF * WF * 2);
  float* c10 = (float*)alloc((size_t)3 * 16 * 16 * 48 * 64 * 4);
  float* c20 = (float*)alloc((size_t)3 * 32 * 8 * 24 * 32 * 4);
  float* c30 = (float*)alloc((size_t)3 * 64 * 4 * 12 * 16 * 4);
  float* c01 = (float*)alloc((size_t)3 * 8 * 32 * 96 * 128 * 4);
  float* c11 = (float*)alloc((size_t)3 * 16 * 16 * 48 * 64 * 4);
  float* c21 = (float*)alloc((size_t)3 * 32 * 8 * 24 * 32 * 4);
  float* c31 = (float*)alloc((size_t)3 * 64 * 4 * 12 * 16 * 4);
  float* d4b = (float*)alloc((size_t)3 * 32 * 8 * 24 * 32 * 4);
  float* d5b = (float*)alloc((size_t)3 * 16 * 16 * 48 * 64 * 4);
  float* d6b = (float*)alloc((size_t)3 * 8 * 32 * 96 * 128 * 4);
  float* fcv = (float*)alloc((size_t)3 * DDEP * HF * WF * 4);
  float* pred = (float*)alloc((size_t)3 * HF * WF * 4);
  float* w1o = (float*)alloc((size_t)3 * 128 * HF * WF * 4);
  float* w2o = (float*)alloc((size_t)3 * 9 * HF * WF * 4);
  if (off > ws_size) return;   // workspace too small: bail deterministically

  auto conv = [&](const void* in, int inHalf,
                  const float* w, const float* g, const float* b, float* o,
                  int N, int Cin, int Cout,
                  int ID, int IH, int IW,
                  int KD, int KH, int KW,
                  int stride, int padD, int padHW,
                  int transposed, int relu,
                  int OD, int OH, int OW) {
    dim3 grid((unsigned)((OD * OH * OW + 16 * NT - 1) / (16 * NT)),
              (unsigned)((Cout + 15) / 16), (unsigned)N);
    conv_wmma_kernel<<<grid, 32, 0, stream>>>(
        in, inHalf, w, g, b, o, Cin, Cout, ID, IH, IW, OD, OH, OW,
        KD, KH, KW, stride, padD, padHW, transposed, relu);
  };

  // ---- 8-layer image conv (shared structure for main + propagation) ----
  auto run_imgconv = [&](int leafBase, float* outFeat) {
    const int cin[8]  = {3, 8, 8, 16, 16, 16, 32, 32};
    const int cout[8] = {8, 8, 16, 16, 16, 32, 32, 32};
    const int kk[8]   = {3, 3, 5, 3, 3, 5, 3, 3};
    const int ss[8]   = {1, 1, 2, 1, 1, 2, 1, 1};
    const void* cur = (const void*)F(IN_IMG);
    int IH = HIMG, IW = WIMG;
    for (int l = 0; l < 8; l++) {
      int pd = kk[l] / 2;
      int OH = (IH + 2 * pd - kk[l]) / ss[l] + 1;
      int OW = (IW + 2 * pd - kk[l]) / ss[l] + 1;
      float* dst = (l == 7) ? outFeat : ((l & 1) ? bufB : bufA);
      conv(cur, 0, F(leafBase + 3 * l), F(leafBase + 3 * l + 1),
           F(leafBase + 3 * l + 2), dst,
           3, cin[l], cout[l], 1, IH, IW, 1, kk[l], kk[l],
           ss[l], 0, pd, 0, 1, 1, OH, OW);
      cur = dst; IH = OH; IW = OW;
    }
  };

  // ================== pipeline ==================
  // 1) main feature extraction -> feat (3, 32, 96, 128)
  run_imgconv(LI_IMG, feat);

  // 2) downsample camera tensors by rs=4
  {
    int tot = V_ * V_ * 3 * HF * WF;
    downsample_kernel<<<(tot + 255) / 256, 256, 0, stream>>>(
        F(IN_BP), bpds, V_ * V_ * 3, HIMG, WIMG, RS, HF, WF);
    downsample_kernel<<<(tot + 255) / 256, 256, 0, stream>>>(
        F(IN_DR), drds, V_ * V_ * 3, HIMG, WIMG, RS, HF, WF);
    int tot2 = V_ * DDEP * HF * WF;
    downsample_kernel<<<(tot2 + 255) / 256, 256, 0, stream>>>(
        F(IN_DEP), ddds, V_ * DDEP, HIMG, WIMG, RS, HF, WF);
  }

  // 3) fused grid-sample + variance cost -> f16 cost (3, 32, 32, 96, 128)
  {
    int tot = V_ * DDEP * HF * WF;
    cost_fuse_kernel<<<(tot + 255) / 256, 256, 0, stream>>>(
        feat, bpds, drds, ddds, cost);
  }

  // 4) 3D UNet on cost volume (all k=3, pad=1)
  conv(cost, 1, F(LI_C1_0), F(LI_C1_0 + 1), F(LI_C1_0 + 2), c10,
       3, 32, 16, 32, 96, 128, 3, 3, 3, 2, 1, 1, 0, 1, 16, 48, 64);
  conv(c10, 0, F(LI_C2_0), F(LI_C2_0 + 1), F(LI_C2_0 + 2), c20,
       3, 16, 32, 16, 48, 64, 3, 3, 3, 2, 1, 1, 0, 1, 8, 24, 32);
  conv(c20, 0, F(LI_C3_0), F(LI_C3_0 + 1), F(LI_C3_0 + 2), c30,
       3, 32, 64, 8, 24, 32, 3, 3, 3, 2, 1, 1, 0, 1, 4, 12, 16);
  conv(cost, 1, F(LI_C0_1), F(LI_C0_1 + 1), F(LI_C0_1 + 2), c01,
       3, 32, 8, 32, 96, 128, 3, 3, 3, 1, 1, 1, 0, 1, 32, 96, 128);
  conv(c10, 0, F(LI_C1_1), F(LI_C1_1 + 1), F(LI_C1_1 + 2), c11,
       3, 16, 16, 16, 48, 64, 3, 3, 3, 1, 1, 1, 0, 1, 16, 48, 64);
  conv(c20, 0, F(LI_C2_1), F(LI_C2_1 + 1), F(LI_C2_1 + 2), c21,
       3, 32, 32, 8, 24, 32, 3, 3, 3, 1, 1, 1, 0, 1, 8, 24, 32);
  conv(c30, 0, F(LI_C3_1), F(LI_C3_1 + 1), F(LI_C3_1 + 2), c31,
       3, 64, 64, 4, 12, 16, 3, 3, 3, 1, 1, 1, 0, 1, 4, 12, 16);
  // decoder (transposed convs, lhs_dilation=2, out = 2*in)
  conv(c31, 0, F(LI_D4), F(LI_D4 + 1), F(LI_D4 + 2), d4b,
       3, 64, 32, 4, 12, 16, 3, 3, 3, 1, 0, 0, 1, 1, 8, 24, 32);
  {
    int n = 3 * 32 * 8 * 24 * 32;
    add_inplace_kernel<<<(n + 255) / 256, 256, 0, stream>>>(d4b, c21, n);
  }
  conv(d4b, 0, F(LI_D5), F(LI_D5 + 1), F(LI_D5 + 2), d5b,
       3, 32, 16, 8, 24, 32, 3, 3, 3, 1, 0, 0, 1, 1, 16, 48, 64);
  {
    int n = 3 * 16 * 16 * 48 * 64;
    add_inplace_kernel<<<(n + 255) / 256, 256, 0, stream>>>(d5b, c11, n);
  }
  conv(d5b, 0, F(LI_D6), F(LI_D6 + 1), F(LI_D6 + 2), d6b,
       3, 16, 8, 16, 48, 64, 3, 3, 3, 1, 0, 0, 1, 1, 32, 96, 128);
  {
    int n = 3 * 8 * 32 * 96 * 128;
    add_inplace_kernel<<<(n + 255) / 256, 256, 0, stream>>>(d6b, c01, n);
  }
  // final 8->1 conv (no BN, no ReLU) -> fcv (3, 32, 96, 128)
  conv(d6b, 0, F(LI_C6_2), nullptr, nullptr, fcv,
       3, 8, 1, 32, 96, 128, 3, 3, 3, 1, 1, 1, 0, 0, 32, 96, 128);

  // 5) softmax depth regression -> pred (3, 96, 128)
  {
    int tot = V_ * HF * WF;
    depth_regress_kernel<<<(tot + 255) / 256, 256, 0, stream>>>(fcv, ddds, pred);
  }

  // 6) propagation net: its own image conv (reuse feat buffer) + w1 + w2
  run_imgconv(LI_PROP_IMG, feat);
  conv(feat, 0, F(LI_W1), F(LI_W1 + 1), F(LI_W1 + 2), w1o,
       3, 32, 128, 1, HF, WF, 1, 3, 3, 1, 0, 1, 0, 1, 1, HF, WF);
  conv(w1o, 0, F(LI_W2), nullptr, nullptr, w2o,
       3, 128, 9, 1, HF, WF, 1, 3, 3, 1, 0, 1, 0, 0, 1, HF, WF);

  // 7) softmax over 9 + neighbor blend -> output (1, 3, 96, 128)
  {
    int tot = V_ * HF * WF;
    prop_combine_kernel<<<(tot + 255) / 256, 256, 0, stream>>>(
        w2o, pred, (float*)d_out);
  }
}
